// CLVPSelfAttention_40329742910173
// MI455X (gfx1250) — compile-verified, matrix operand order checked by reading, not run on hardware
//
#include <hip/hip_runtime.h>
#include <math.h>

// CLVP self-attention for MI455X (gfx1250, wave32, WMMA bf16 16x16x32 + TDM).
// B=4, S=1024, E=1024, H=16, D=64, ROT=32.
// f32->bf16 convert -> QKV GEMMs (TDM-staged, bf16 WMMA) -> RoPE(q,k,v) ->
// causal flash attention (TDM K-tiles, bf16 WMMA) -> out projection (+bias, f32).

#define Bn 4
#define Sn 1024
#define En 1024
#define Hn 16
#define Dn 64
#define ROTn 32
#define NEGINF (-3.0e38f)

typedef __attribute__((ext_vector_type(16))) __bf16 v16bf;
typedef __attribute__((ext_vector_type(8)))  __bf16 v8bf;
typedef __attribute__((ext_vector_type(8)))  float  v8f;
typedef unsigned int u32x4 __attribute__((ext_vector_type(4)));
typedef int          i32x4 __attribute__((ext_vector_type(4)));
typedef int          i32x8 __attribute__((ext_vector_type(8)));

static __device__ __forceinline__ __bf16 f2bf(float f) {
    unsigned int u = __float_as_uint(f);
    unsigned int r = (u + 0x7FFFu + ((u >> 16) & 1u)) >> 16;
    unsigned short s = (unsigned short)r;
    __bf16 o; __builtin_memcpy(&o, &s, 2); return o;
}
static __device__ __forceinline__ float bf2f(__bf16 b) {
    unsigned short s; __builtin_memcpy(&s, &b, 2);
    return __uint_as_float(((unsigned int)s) << 16);
}
static __device__ __forceinline__ v8f zero8() {
    v8f z;
#pragma unroll
    for (int i = 0; i < 8; ++i) z[i] = 0.f;
    return z;
}
static __device__ __forceinline__ v16bf ld16(const __bf16* p) {
    v8bf lo = *(const v8bf*)(p);
    v8bf hi = *(const v8bf*)(p + 8);
    v16bf r;
    __builtin_memcpy(&r, &lo, 16);
    __builtin_memcpy(((char*)&r) + 16, &hi, 16);
    return r;
}
static __device__ __forceinline__ v8f wmma_bf16(v16bf a, v16bf b, v8f c) {
    return __builtin_amdgcn_wmma_f32_16x16x32_bf16(false, a, false, b, (short)0, c, false, false);
}

// LDS byte address = low 32 bits of the flat address (LDS aperture keeps the
// offset in addr[31:0]; high bits are the shared-aperture tag).
static __device__ __forceinline__ unsigned lds_addr_of(const void* p) {
    return (unsigned)(uintptr_t)p;
}

// Issue one TDM tensor_load_to_lds of a 2D bf16 tile:
//   tile_d1 rows x tile_d0 elements, global row stride `stride` elements,
//   LDS destination padded by `pad_amount_dw` DWORDs every `pad_interval_code`
//   (log2(DWORDs)-1) DWORDs. Tracked on TENSORcnt (wave-level; EXEC ignored).
static __device__ __forceinline__ void tdm_load_tile(unsigned lds_byte_addr,
                                                     const __bf16* gptr,
                                                     unsigned tensor_d0, unsigned tensor_d1,
                                                     unsigned stride_elems,
                                                     unsigned tile_d0, unsigned tile_d1,
                                                     unsigned pad_interval_code,
                                                     unsigned pad_amount_code) {
    unsigned long long ga = (unsigned long long)(uintptr_t)gptr;
    u32x4 g0;
    g0[0] = 1u;                                              // count=1, user-mode load
    g0[1] = lds_byte_addr;                                   // lds_addr
    g0[2] = (unsigned)(ga & 0xffffffffu);                    // global_addr[31:0]
    g0[3] = (unsigned)((ga >> 32) & 0x01ffffffu) | (2u << 30); // global_addr[56:32] | type=2
    i32x8 g1;
    g1[0] = (int)((1u << 16)                                 // data_size = 2 bytes
                | (1u << 20)                                 // pad_enable
                | (pad_interval_code << 22)
                | (pad_amount_code << 25));
    g1[1] = (int)((tensor_d0 & 0xffffu) << 16);              // tensor_dim0[15:0]
    g1[2] = (int)(((tensor_d0 >> 16) & 0xffffu) | ((tensor_d1 & 0xffffu) << 16));
    g1[3] = (int)(((tensor_d1 >> 16) & 0xffffu) | ((tile_d0 & 0xffffu) << 16));
    g1[4] = (int)(tile_d1 & 0xffffu);                        // tile_dim1 (tile_dim2 = 0)
    g1[5] = (int)stride_elems;                               // tensor_dim0_stride[31:0]
    g1[6] = 0;
    g1[7] = 0;
    i32x4 z4 = {0, 0, 0, 0};
#if __clang_major__ >= 23
    i32x8 z8 = {0, 0, 0, 0, 0, 0, 0, 0};
    __builtin_amdgcn_tensor_load_to_lds(g0, g1, z4, z4, z8, 0);
#else
    __builtin_amdgcn_tensor_load_to_lds(g0, g1, z4, z4, 0);
#endif
}

// ---------------------------------------------------------------- convert
__global__ __launch_bounds__(256) void cvt_bf16_kernel(const float* __restrict__ in,
                                                       __bf16* __restrict__ out, int n) {
    for (int i = blockIdx.x * 256 + threadIdx.x; i < n; i += gridDim.x * 256)
        out[i] = f2bf(in[i]);
}

// ---------------------------------------------------------------- GEMM: C = A * W^T
// Block tile 128x128, 256 threads = 8 waves (4 row-groups x 2 col-groups),
// each wave 32x64 = 2x4 WMMA tiles, K-step 32. Double-buffered LDS tiles filled
// by the Tensor Data Mover (wave 0 issues; TENSORcnt + barrier synchronize);
// TDM applies the 4-DWORD row padding (32 bf16 = 16 DW rows -> interval code 3).
#define LDA 40 // 32-element rows padded to 40 (80B stride, conflict-free)

__global__ __launch_bounds__(256) void gemm_qkv_kernel(const __bf16* __restrict__ A,
                                                       const __bf16* __restrict__ W,
                                                       __bf16* __restrict__ C, float scale) {
    __shared__ __bf16 Alds[2][128 * LDA];
    __shared__ __bf16 Blds[2][128 * LDA];
    const int tid = threadIdx.x, lane = tid & 31, wave = tid >> 5;
    const int wr = wave & 3, wc = wave >> 2;
    const int m0 = blockIdx.y * 128, n0 = blockIdx.x * 128;
    const int ln = lane & 15, lh = lane >> 4;

    v8f acc[2][4];
#pragma unroll
    for (int t = 0; t < 2; ++t)
#pragma unroll
        for (int j = 0; j < 4; ++j) acc[t][j] = zero8();

    const __bf16* gA = &A[(size_t)m0 * En];
    const __bf16* gW = &W[(size_t)n0 * En];
    const unsigned aAddr[2] = {lds_addr_of(&Alds[0][0]), lds_addr_of(&Alds[1][0])};
    const unsigned bAddr[2] = {lds_addr_of(&Blds[0][0]), lds_addr_of(&Blds[1][0])};

    if (wave == 0) { // prologue: fill buffer 0 (k0 = 0)
        tdm_load_tile(aAddr[0], gA, En, Bn * Sn, En, 32, 128, 3, 3);
        tdm_load_tile(bAddr[0], gW, En, En, En, 32, 128, 3, 3);
    }

    for (int k0 = 0, it = 0; k0 < En; k0 += 32, ++it) {
        const int cur = it & 1;
        if (wave == 0) {
            if (k0 + 32 < En) { // DMA next tile into the other buffer
                tdm_load_tile(aAddr[cur ^ 1], gA + k0 + 32, En, Bn * Sn, En, 32, 128, 3, 3);
                tdm_load_tile(bAddr[cur ^ 1], gW + k0 + 32, En, En, En, 32, 128, 3, 3);
                __builtin_amdgcn_s_wait_tensorcnt((short)2); // current pair resident
            } else {
                __builtin_amdgcn_s_wait_tensorcnt((short)0);
            }
        }
        __syncthreads(); // buffer `cur` ready for all waves

        const __bf16* Ab = Alds[cur];
        const __bf16* Bb = Blds[cur];
        v16bf a0 = ld16(&Ab[(wr * 32 +      ln) * LDA + lh * 16]);
        v16bf a1 = ld16(&Ab[(wr * 32 + 16 + ln) * LDA + lh * 16]);
#pragma unroll
        for (int j = 0; j < 4; ++j) {
            v16bf bj = ld16(&Bb[(wc * 64 + j * 16 + ln) * LDA + lh * 16]);
            acc[0][j] = wmma_bf16(a0, bj, acc[0][j]);
            acc[1][j] = wmma_bf16(a1, bj, acc[1][j]);
        }
        __syncthreads(); // all reads of `cur` done before TDM may overwrite it
    }
#pragma unroll
    for (int t = 0; t < 2; ++t)
#pragma unroll
        for (int j = 0; j < 4; ++j)
#pragma unroll
            for (int r = 0; r < 8; ++r) {
                int row = m0 + wr * 32 + t * 16 + r + 8 * lh;
                int col = n0 + wc * 64 + j * 16 + ln;
                C[(size_t)row * En + col] = f2bf(acc[t][j][r] * scale);
            }
}

__global__ __launch_bounds__(256) void gemm_out_kernel(const __bf16* __restrict__ A,
                                                       const __bf16* __restrict__ W,
                                                       const float* __restrict__ bias,
                                                       float* __restrict__ C) {
    __shared__ __bf16 Alds[2][128 * LDA];
    __shared__ __bf16 Blds[2][128 * LDA];
    const int tid = threadIdx.x, lane = tid & 31, wave = tid >> 5;
    const int wr = wave & 3, wc = wave >> 2;
    const int m0 = blockIdx.y * 128, n0 = blockIdx.x * 128;
    const int ln = lane & 15, lh = lane >> 4;

    v8f acc[2][4];
#pragma unroll
    for (int t = 0; t < 2; ++t)
#pragma unroll
        for (int j = 0; j < 4; ++j) acc[t][j] = zero8();

    const __bf16* gA = &A[(size_t)m0 * En];
    const __bf16* gW = &W[(size_t)n0 * En];
    const unsigned aAddr[2] = {lds_addr_of(&Alds[0][0]), lds_addr_of(&Alds[1][0])};
    const unsigned bAddr[2] = {lds_addr_of(&Blds[0][0]), lds_addr_of(&Blds[1][0])};

    if (wave == 0) {
        tdm_load_tile(aAddr[0], gA, En, Bn * Sn, En, 32, 128, 3, 3);
        tdm_load_tile(bAddr[0], gW, En, En, En, 32, 128, 3, 3);
    }

    for (int k0 = 0, it = 0; k0 < En; k0 += 32, ++it) {
        const int cur = it & 1;
        if (wave == 0) {
            if (k0 + 32 < En) {
                tdm_load_tile(aAddr[cur ^ 1], gA + k0 + 32, En, Bn * Sn, En, 32, 128, 3, 3);
                tdm_load_tile(bAddr[cur ^ 1], gW + k0 + 32, En, En, En, 32, 128, 3, 3);
                __builtin_amdgcn_s_wait_tensorcnt((short)2);
            } else {
                __builtin_amdgcn_s_wait_tensorcnt((short)0);
            }
        }
        __syncthreads();

        const __bf16* Ab = Alds[cur];
        const __bf16* Bb = Blds[cur];
        v16bf a0 = ld16(&Ab[(wr * 32 +      ln) * LDA + lh * 16]);
        v16bf a1 = ld16(&Ab[(wr * 32 + 16 + ln) * LDA + lh * 16]);
#pragma unroll
        for (int j = 0; j < 4; ++j) {
            v16bf bj = ld16(&Bb[(wc * 64 + j * 16 + ln) * LDA + lh * 16]);
            acc[0][j] = wmma_bf16(a0, bj, acc[0][j]);
            acc[1][j] = wmma_bf16(a1, bj, acc[1][j]);
        }
        __syncthreads();
    }
#pragma unroll
    for (int t = 0; t < 2; ++t)
#pragma unroll
        for (int j = 0; j < 4; ++j)
#pragma unroll
            for (int r = 0; r < 8; ++r) {
                int row = m0 + wr * 32 + t * 16 + r + 8 * lh;
                int col = n0 + wc * 64 + j * 16 + ln;
                C[(size_t)row * En + col] = acc[t][j][r] + bias[col];
            }
}

// ---------------------------------------------------------------- RoPE (in-place on bf16)
__global__ __launch_bounds__(128) void rope_kernel(__bf16* __restrict__ Q,
                                                   __bf16* __restrict__ Kb,
                                                   __bf16* __restrict__ V,
                                                   const float* __restrict__ freqs) {
    const int lane = threadIdx.x & 31;
    const int grp = blockIdx.x * 4 + (threadIdx.x >> 5); // token*H + h
    const int h = grp & (Hn - 1);
    const int token = grp >> 4;
    const int s = token & (Sn - 1);
    __bf16* base = (blockIdx.y == 0) ? Q : (blockIdx.y == 1) ? Kb : V;
    __bf16* x = base + (size_t)token * En + h * Dn;
    float f = freqs[s * ROTn + lane];
    float xv = bf2f(x[lane]);
    int partner = (lane < 16) ? lane + 16 : lane - 16;
    float xp = bf2f(x[partner]);
    float rot = (lane < 16) ? -xp : xp;
    float o = xv * __cosf(f) + rot * __sinf(f);
    x[lane] = f2bf(o); // store issued after all lane loads (single wave instruction stream)
}

// ---------------------------------------------------------------- causal flash attention
// Grid: (S/64, B*H). Block 128 = 4 waves; wave w owns Q rows [q0+16w, q0+16w+16).
// K tile (32x64) DMA'd by TDM with 4-DWORD row padding (64 bf16 = 32 DW -> code 4);
// V staged transposed by the block (TDM has no transpose).
#define KP 72
#define VP 40
__global__ __launch_bounds__(128) void flash_attn_kernel(const __bf16* __restrict__ Q,
                                                         const __bf16* __restrict__ Kb,
                                                         const __bf16* __restrict__ V,
                                                         __bf16* __restrict__ O) {
    __shared__ __bf16 Klds[32 * KP];
    __shared__ __bf16 Vlds[64 * VP];     // transposed: [d][key]
    __shared__ __bf16 Plds[4 * 16 * VP]; // per-wave 16x32 P tile

    const int tid = threadIdx.x, lane = tid & 31, wave = tid >> 5;
    const int ln = lane & 15, lh = lane >> 4;
    const int b = blockIdx.y >> 4, h = blockIdx.y & 15;
    const int q0 = blockIdx.x * 64;
    const int qrow = q0 + wave * 16;
    const size_t base = ((size_t)b * Sn) * En + h * Dn;
    const unsigned kAddr = lds_addr_of(&Klds[0]);

    v16bf aQ0 = ld16(&Q[base + (size_t)(qrow + ln) * En + lh * 16]);
    v16bf aQ1 = ld16(&Q[base + (size_t)(qrow + ln) * En + lh * 16 + 32]);

    v8f acc[4];
#pragma unroll
    for (int j = 0; j < 4; ++j) acc[j] = zero8();
    float m_i[8], l_i[8];
#pragma unroll
    for (int r = 0; r < 8; ++r) { m_i[r] = NEGINF; l_i[r] = 0.f; }

    const int krow = tid >> 2;       // key within block: 0..31
    const int kcol = (tid & 3) * 16; // d chunk
    const int kvend = q0 + 64;

    for (int kb = 0; kb < kvend; kb += 32) {
        __syncthreads();
        if (wave == 0) // DMA K tile: 32 rows x 64 bf16, row stride E, padded to KP
            tdm_load_tile(kAddr, &Kb[base + (size_t)kb * En], En, Bn * Sn, En, 64, 32, 4, 3);
        { // stage V transposed
            const __bf16* gv = &V[base + (size_t)(kb + krow) * En + kcol];
            v8bf v0v = *(const v8bf*)(gv);
            v8bf v1v = *(const v8bf*)(gv + 8);
#pragma unroll
            for (int e = 0; e < 8; ++e) Vlds[(kcol + e) * VP + krow] = v0v[e];
#pragma unroll
            for (int e = 0; e < 8; ++e) Vlds[(kcol + 8 + e) * VP + krow] = v1v[e];
            if (kb + 32 < kvend) __builtin_prefetch(gv + 32 * En, 0, 1);
        }
        if (wave == 0) __builtin_amdgcn_s_wait_tensorcnt((short)0);
        __syncthreads();

        v8f s0 = zero8(), s1 = zero8();
        v16bf bk;
        bk = ld16(&Klds[(     ln) * KP +      lh * 16]); s0 = wmma_bf16(aQ0, bk, s0);
        bk = ld16(&Klds[(     ln) * KP + 32 + lh * 16]); s0 = wmma_bf16(aQ1, bk, s0);
        bk = ld16(&Klds[(16 + ln) * KP +      lh * 16]); s1 = wmma_bf16(aQ0, bk, s1);
        bk = ld16(&Klds[(16 + ln) * KP + 32 + lh * 16]); s1 = wmma_bf16(aQ1, bk, s1);

#pragma unroll
        for (int r = 0; r < 8; ++r) {
            int row = qrow + r + 8 * lh;
            int key0 = kb + ln, key1 = key0 + 16;
            float v0 = (key0 <= row) ? s0[r] : NEGINF;
            float v1 = (key1 <= row) ? s1[r] : NEGINF;
            float mx = fmaxf(v0, v1);
#pragma unroll
            for (int d = 1; d < 16; d <<= 1) mx = fmaxf(mx, __shfl_xor(mx, d, 32));
            float mnew = fmaxf(m_i[r], mx);
            float corr = __expf(m_i[r] - mnew);
            m_i[r] = mnew;
            float p0 = __expf(v0 - mnew);
            float p1 = __expf(v1 - mnew);
            float ps = p0 + p1;
#pragma unroll
            for (int d = 1; d < 16; d <<= 1) ps += __shfl_xor(ps, d, 32);
            l_i[r] = l_i[r] * corr + ps;
#pragma unroll
            for (int j = 0; j < 4; ++j) acc[j][r] *= corr;
            int prow = r + 8 * lh;
            Plds[(wave * 16 + prow) * VP + ln]      = f2bf(p0);
            Plds[(wave * 16 + prow) * VP + 16 + ln] = f2bf(p1);
        }

        v16bf aP = ld16(&Plds[(wave * 16 + ln) * VP + lh * 16]);
#pragma unroll
        for (int j = 0; j < 4; ++j) {
            v16bf bv = ld16(&Vlds[(j * 16 + ln) * VP + lh * 16]);
            acc[j] = wmma_bf16(aP, bv, acc[j]);
        }
    }

#pragma unroll
    for (int j = 0; j < 4; ++j)
#pragma unroll
        for (int r = 0; r < 8; ++r) {
            int row = qrow + r + 8 * lh;
            int col = h * Dn + j * 16 + ln;
            O[((size_t)b * Sn + row) * En + col] = f2bf(acc[j][r] / l_i[r]);
        }
}

// ---------------------------------------------------------------- launch
extern "C" void kernel_launch(void* const* d_in, const int* in_sizes, int n_in,
                              void* d_out, int out_size, void* d_ws, size_t ws_size,
                              hipStream_t stream) {
    const float* hs  = (const float*)d_in[0];
    const float* rot = (const float*)d_in[1];
    const float* q_w = (const float*)d_in[2];
    const float* k_w = (const float*)d_in[3];
    const float* v_w = (const float*)d_in[4];
    const float* o_w = (const float*)d_in[5];
    const float* o_b = (const float*)d_in[6];
    float* out = (float*)d_out;

    const int M = Bn * Sn;  // 4096 tokens
    const int EE = En * En;
    const int ME = M * En;

    __bf16* hsb = (__bf16*)d_ws;
    __bf16* qwb = hsb + ME;
    __bf16* kwb = qwb + EE;
    __bf16* vwb = kwb + EE;
    __bf16* owb = vwb + EE;
    __bf16* Qb  = owb + EE;
    __bf16* Kb  = Qb + ME;
    __bf16* Vb  = Kb + ME;
    __bf16* AOb = Vb + ME;

    cvt_bf16_kernel<<<dim3(2048), 256, 0, stream>>>(hs, hsb, ME);
    cvt_bf16_kernel<<<dim3(1024), 256, 0, stream>>>(q_w, qwb, EE);
    cvt_bf16_kernel<<<dim3(1024), 256, 0, stream>>>(k_w, kwb, EE);
    cvt_bf16_kernel<<<dim3(1024), 256, 0, stream>>>(v_w, vwb, EE);
    cvt_bf16_kernel<<<dim3(1024), 256, 0, stream>>>(o_w, owb, EE);

    dim3 ggrid(En / 128, M / 128); // (8, 32)
    gemm_qkv_kernel<<<ggrid, 256, 0, stream>>>(hsb, qwb, Qb, 0.125f); // D^-0.5
    gemm_qkv_kernel<<<ggrid, 256, 0, stream>>>(hsb, kwb, Kb, 1.0f);
    gemm_qkv_kernel<<<ggrid, 256, 0, stream>>>(hsb, vwb, Vb, 1.0f);

    rope_kernel<<<dim3((M * Hn) / 4, 3), 128, 0, stream>>>(Qb, Kb, Vb, rot);

    flash_attn_kernel<<<dim3(Sn / 64, Bn * Hn), 128, 0, stream>>>(Qb, Kb, Vb, AOb);

    gemm_out_kernel<<<ggrid, 256, 0, stream>>>(AOb, owb, o_b, out);
}
